// AVNNType1Conv2d_6571299963114
// MI455X (gfx1250) — matrix operand order repaired and checked by reading, not compiled, not run on hardware
//
#include <hip/hip_runtime.h>

// ---------------------------------------------------------------------------
// AVNN Type-1 Conv2d fused kernel for gfx1250 (MI455X, wave32, WMMA).
//
// Path 1: act_out = relu(conv3x3(act) + bias)      -> implicit GEMM via WMMA
// Path 2: carry   = patchsum(x*y)/(patchsum(x)+eps); carry_out = 1x1 proj
//
// Shapes: B=16, C=O=64, H=W=128, Hp=Wp=126, K=3.
// Block: 256 threads (8 waves), computes 64 ochan x 32 px x 3 rows.
// Each A fragment load feeds 3 WMMAs (row reuse) to cut L2 weight traffic 3x.
// ---------------------------------------------------------------------------

typedef __attribute__((ext_vector_type(16))) _Float16 v16h;
typedef __attribute__((ext_vector_type(8)))  _Float16 v8h;
typedef __attribute__((ext_vector_type(8)))  float    v8f;

#define BATCH 16
#define CCH   64
#define OCH   64
#define HIN   128
#define WIN   128
#define HP    126
#define WP    126
#define EPSV  1e-6f

#define ROWS  3                        // output rows per block
#define IROWS (ROWS + 2)               // input rows per block

// K-ordering for the conv GEMM: K = tap*64 + c, tap = kh*3+kw, Ktot = 576
#define CONV_PACK_HALVES 36864         // [mb(4)][s(18)][lane(32)] x 16
#define PROJ_PACK_HALVES 4096          // [mb(4)][s(2)][lane(32)]  x 16

// LDS act/xy tile layout: [row(5)][col(34)][c], channel stride padded to 72
// (stride 144B = 36 banks -> conflict-free-ish, 16B aligned for b128 loads)
#define CSTR   72
#define TILE_W 34
#define LDS_TILE (IROWS * TILE_W * CSTR)   // 12240 halves

// ---------------------------------------------------------------------------
// Kernel 0: pack f32 weights into f16 WMMA A-fragment layout in workspace.
// A fragment k order used here (contiguous; B loads match): element e of lane l holds
//   K = 32*s + e + (l>=16 ? 16 : 0),  M = mb*16 + (l & 15)
// ---------------------------------------------------------------------------
__global__ void avnn_pack_weights(const float* __restrict__ conv_w,
                                  const float* __restrict__ proj_w,
                                  _Float16* __restrict__ ws) {
  int idx = blockIdx.x * 256 + threadIdx.x;
  if (idx < CONV_PACK_HALVES) {
    int e    = idx & 15;
    int lane = (idx >> 4) & 31;
    int s    = (idx >> 9) % 18;
    int mb   = (idx >> 9) / 18;
    int hi   = lane >> 4;
    int koff = e + hi * 16;
    int K    = 32 * s + koff;          // 0..575
    int t    = K >> 6;                 // tap 0..8
    int c    = K & 63;                 // channel
    int kh   = t / 3, kw = t % 3;
    int M    = mb * 16 + (lane & 15);  // out channel
    ws[idx]  = (_Float16)conv_w[(((size_t)M * CCH + c) * 3 + kh) * 3 + kw];
  } else if (idx < CONV_PACK_HALVES + PROJ_PACK_HALVES) {
    int j    = idx - CONV_PACK_HALVES;
    int e    = j & 15;
    int lane = (j >> 4) & 31;
    int s    = (j >> 9) & 1;
    int mb   = j >> 10;
    int hi   = lane >> 4;
    int koff = e + hi * 16;
    int c    = 32 * s + koff;          // 0..63
    int M    = mb * 16 + (lane & 15);
    ws[idx]  = (_Float16)proj_w[(size_t)M * CCH + c];
  }
}

// 16-byte-aligned LDS load of 16 halves (two v8h -> v16h).
__device__ __forceinline__ v16h ld16h(const _Float16* p) {
  v8h lo = *(const v8h*)p;
  v8h hi = *(const v8h*)(p + 8);
  return __builtin_shufflevector(lo, hi, 0,1,2,3,4,5,6,7,8,9,10,11,12,13,14,15);
}

// ---------------------------------------------------------------------------
// Main fused kernel. Grid: (4 w-strips, 42 row-triples, 16 batches).
// Wave w: mb = w>>1 (16-channel block), g = w&1 (16-pixel column group);
// each wave accumulates 3 output rows per A-fragment load.
// ---------------------------------------------------------------------------
__global__ __launch_bounds__(256)
void avnn_fused_kernel(const float* __restrict__ in,     // [B,C,H,W,2]
                       const float* __restrict__ conv_b, // [O]
                       const float* __restrict__ proj_b, // [O]
                       const _Float16* __restrict__ ws,  // packed weights
                       float* __restrict__ out) {        // [B,O,HP,WP,2]
  __shared__ _Float16 sAct[LDS_TILE];          // f16 activator [r][col][c]
  __shared__ _Float16 sXY [LDS_TILE];          // f16 x*y       [r][col][c]
  __shared__ _Float16 sCar[ROWS * 32 * CSTR];  // f16 carry     [rr][px][c]

  const int tid = threadIdx.x;
  const int w0  = blockIdx.x * 32;     // strip base: 0,32,64,96
  const int h0  = blockIdx.y * ROWS;   // 0,3,...,123
  const int b   = blockIdx.z;          // 0..15

  // ---- Phase A: load 64ch x 5row x 34col patch (act,carry) -> LDS f16 ----
  const int NTILE = CCH * IROWS * TILE_W;      // 10880 elements
  for (int e = tid; e < NTILE; e += 256) {
    int c    = e / (IROWS * TILE_W);
    int rem  = e - c * (IROWS * TILE_W);
    int r    = rem / TILE_W;
    int wcol = rem - r * TILE_W;
    int wg   = w0 + wcol;
    if (wg > WIN - 1) wg = WIN - 1;            // clamp; only feeds dead lanes
    const float2 v = *(const float2*)
        (in + ((((size_t)b * CCH + c) * HIN + (h0 + r)) * WIN + wg) * 2);
    int li = (r * TILE_W + wcol) * CSTR + c;
    sAct[li] = (_Float16)v.x;
    sXY [li] = (_Float16)(v.x * v.y);
  }
  // Prefetch next row-triple's first rows (global_prefetch_b8).
  {
    int hn = (h0 + IROWS < HIN) ? h0 + IROWS : HIN - 1;
    __builtin_prefetch(in + ((((size_t)b * CCH) * HIN + hn) * WIN + w0) * 2,
                       0, 1);
  }
  __syncthreads();

  // ---- Phase B: depthwise 3x3 sums + divide -> carry f16 [rr][px][c] ------
  // Sliding-window column sums: 768 tasks of (c, rr, 8-px strip).
  for (int task = tid; task < CCH * ROWS * 4; task += 256) {
    int c     = task / (ROWS * 4);
    int rem   = task - c * (ROWS * 4);
    int rr    = rem >> 2;
    int j0    = (rem & 3) * 8;
    float colx[10], colq[10];
#pragma unroll
    for (int q = 0; q < 10; ++q) {
      float ax = 0.f, qx = 0.f;
#pragma unroll
      for (int r = 0; r < 3; ++r) {
        int li = ((rr + r) * TILE_W + (j0 + q)) * CSTR + c;
        ax += (float)sAct[li];
        qx += (float)sXY [li];
      }
      colx[q] = ax;
      colq[q] = qx;
    }
#pragma unroll
    for (int j = 0; j < 8; ++j) {
      float sx  = colx[j] + colx[j + 1] + colx[j + 2];
      float sxy = colq[j] + colq[j + 1] + colq[j + 2];
      sCar[(rr * 32 + j0 + j) * CSTR + c] = (_Float16)(sxy / (sx + EPSV));
    }
  }
  __syncthreads();

  // ---- Phase C: WMMA GEMMs (EXEC all-ones; no divergence here) ------------
  const int wave = tid >> 5;
  const int lane = tid & 31;
  const int mb   = wave >> 1;        // out-channel block 0..3
  const int g    = wave & 1;         // pixel column group 0..1
  const int ln   = lane & 15;
  const int hi   = lane >> 4;

  const v16h* packA = (const v16h*)ws;                        // conv A frags
  const v16h* packP = (const v16h*)(ws + CONV_PACK_HALVES);   // proj A frags

  // conv3x3 implicit GEMM: 18 K-steps, each A fragment reused for 3 rows.
  v8f acc[ROWS] = {};
#pragma unroll
  for (int s = 0; s < 18; ++s) {
    v16h a = packA[(mb * 18 + s) * 32 + lane];
    int t  = s >> 1;                       // tap 0..8
    int kh = t / 3, kw = t - kh * 3;
    int c0 = ((s & 1) << 5) + (hi << 4);   // 16 contiguous channels
    int col = g * 16 + ln + kw;            // B column = pixel (+kw shift)
#pragma unroll
    for (int rr = 0; rr < ROWS; ++rr) {
      v16h bf = ld16h(&sAct[((rr + kh) * TILE_W + col) * CSTR + c0]);
      acc[rr] = __builtin_amdgcn_wmma_f32_16x16x32_f16(
          false, a, false, bf, (short)0, acc[rr], false, false);
    }
  }

  // 1x1 projection GEMM on carry: 2 K-steps x 3 rows.
  v8f pacc[ROWS] = {};
#pragma unroll
  for (int s = 0; s < 2; ++s) {
    v16h a = packP[(mb * 2 + s) * 32 + lane];
    int c0 = (s << 5) + (hi << 4);
#pragma unroll
    for (int rr = 0; rr < ROWS; ++rr) {
      v16h bf = ld16h(&sCar[(rr * 32 + g * 16 + ln) * CSTR + c0]);
      pacc[rr] = __builtin_amdgcn_wmma_f32_16x16x32_f16(
          false, a, false, bf, (short)0, pacc[rr], false, false);
    }
  }

  // ---- Phase D: bias + relu, interleaved float2 store ---------------------
  const int wpix = w0 + g * 16 + ln;
  if (wpix < WP) {
    float cb8[8], pb8[8];
#pragma unroll
    for (int r = 0; r < 8; ++r) {
      int o  = mb * 16 + hi * 8 + r;     // C/D layout: lanes>=16 hold M+8
      cb8[r] = conv_b[o];
      pb8[r] = proj_b[o];
    }
#pragma unroll
    for (int rr = 0; rr < ROWS; ++rr) {
      int h = h0 + rr;
#pragma unroll
      for (int r = 0; r < 8; ++r) {
        int o = mb * 16 + hi * 8 + r;
        float a_out = acc[rr][r] + cb8[r];
        a_out = a_out > 0.f ? a_out : 0.f;
        float c_out = pacc[rr][r] + pb8[r];
        size_t oi = ((((size_t)b * OCH + o) * HP + h) * WP + wpix) * 2;
        *(float2*)(out + oi) = make_float2(a_out, c_out);  // b64 store
      }
    }
  }
}

// ---------------------------------------------------------------------------
extern "C" void kernel_launch(void* const* d_in, const int* in_sizes, int n_in,
                              void* d_out, int out_size, void* d_ws, size_t ws_size,
                              hipStream_t stream) {
  const float* in     = (const float*)d_in[0];   // [16,64,128,128,2]
  const float* conv_w = (const float*)d_in[1];   // [64,64,3,3]
  const float* conv_b = (const float*)d_in[2];   // [64]
  const float* proj_w = (const float*)d_in[3];   // [64,64,1,1]
  const float* proj_b = (const float*)d_in[4];   // [64]
  float*       outp   = (float*)d_out;
  _Float16*    wsp    = (_Float16*)d_ws;         // 81920 bytes used

  // Pack weights into WMMA fragment layout (40960 halves total).
  avnn_pack_weights<<<160, 256, 0, stream>>>(conv_w, proj_w, wsp);

  // Fused conv + carry kernel: 3 output rows per block.
  dim3 grid(4, HP / ROWS, BATCH);
  avnn_fused_kernel<<<grid, 256, 0, stream>>>(in, conv_b, proj_b, wsp, outp);
}
